// FGNNHG_78529182040869
// MI455X (gfx1250) — matrix operand order repaired and checked by reading, not compiled, not run on hardware
//
#include <hip/hip_runtime.h>
#include <hip/hip_bf16.h>
#include <math.h>

// ============================================================================
// FGNNHG forward for MI455X (gfx1250, wave32).
// Dense GEMMs use v_wmma_f32_16x16x32_bf16 (fp32 accumulate, bf16 operands
// converted on the LDS staging path). Graph message passing uses fp32 atomics.
// ============================================================================

typedef __attribute__((ext_vector_type(16))) __bf16 v16bf;
typedef __attribute__((ext_vector_type(8)))  float  v8f;

#define G_N   12000
#define D_N   6000
#define HID   512
#define OUTC  256
#define E_GG  384000
#define E_DD  192000
#define E_DG  300000
#define E_GD  300000
#define NPAIR 16384

// ---------------------------------------------------------------------------
// Block reduction (blockDim.x == 256)
// ---------------------------------------------------------------------------
__device__ __forceinline__ float blockReduceSum256(float v) {
    __shared__ float red[256];
    int tid = threadIdx.x;
    red[tid] = v;
    __syncthreads();
    for (int s = 128; s > 0; s >>= 1) {
        if (tid < s) red[tid] += red[tid + s];
        __syncthreads();
    }
    float r = red[0];
    __syncthreads();
    return r;
}

__device__ __forceinline__ void atomicMaxF(float* addr, float val) {
    unsigned int* ua = (unsigned int*)addr;
    unsigned int old = *ua;
    while (true) {
        float f = __uint_as_float(old);
        if (f >= val) break;
        unsigned int assumed = old;
        old = atomicCAS(ua, assumed, __float_as_uint(val));
        if (old == assumed) break;
    }
}

// ---------------------------------------------------------------------------
// GEMM: C[M,N] = A[M,K](lda) @ B[K,N](ldb) + bias[N]   (row-major, fp32 I/O)
// Block tile 64x128, BK=32, 8 waves in a 2x4 grid; each wave owns a 32x32 C
// tile -> 4 x v_wmma_f32_16x16x32_bf16 per K-step with A-fragment reuse.
//
// LDS layouts chosen so every lane's 16 fragment values are two contiguous
// 16B chunks (ds_load_b128):
//   sA [m][k]   : A frag lane needs K {ak..ak+7} and {ak+16..ak+23} of row m
//   sBt[n][k]   : B tile stored transposed; lane needs K {bk..bk+7} and {bk+16..bk+23} of col n
// Fragment layouts per CDNA5 ISA 7.12.2 (wave32):
//   A 16x32 bf16 : lanes 0-15 row M=lane, K={0..7,16..23}; lanes 16-31 K={8..15,24..31}
//   B 32x16 bf16 : lanes 0-15 col N=lane, K={0..7,16..23}; lanes 16-31 K={8..15,24..31}
//   C 16x16 f32  : vgpr j -> (M = j + (lane<16?0:8), N = lane&15)
// ---------------------------------------------------------------------------
#define BM 64
#define BN 128
#define BK 32

__global__ __launch_bounds__(256) void gemm_bf16_wmma(
    const float* __restrict__ A, int lda,
    const float* __restrict__ B, int ldb,
    const float* __restrict__ bias,
    float* __restrict__ C, int ldc,
    int M, int N, int K)
{
    __shared__ __align__(16) __bf16 sA[BM][BK];    // 4 KB
    __shared__ __align__(16) __bf16 sBt[BN][BK];   // 8 KB (transposed tile)

    const int tid  = threadIdx.x;
    const int lane = tid & 31;
    const int wave = tid >> 5;
    const int wm   = wave >> 2;          // 0..1 : 32-row strip
    const int wn   = wave & 3;           // 0..3 : 32-col strip
    const int m0   = blockIdx.y * BM;
    const int n0   = blockIdx.x * BN;

    v8f acc00 = {0.f,0.f,0.f,0.f,0.f,0.f,0.f,0.f};
    v8f acc01 = {0.f,0.f,0.f,0.f,0.f,0.f,0.f,0.f};
    v8f acc10 = {0.f,0.f,0.f,0.f,0.f,0.f,0.f,0.f};
    v8f acc11 = {0.f,0.f,0.f,0.f,0.f,0.f,0.f,0.f};

    const bool mFull = (m0 + BM <= M);
    const bool nFull = (n0 + BN <= N);

    for (int k0 = 0; k0 < K; k0 += BK) {
        const bool kFull = (k0 + BK <= K);

        // ---- stage A tile (64x32) ----
        if (mFull && kFull) {
            // thread -> row tid>>2, 8 consecutive K; two float4 loads, one b128 store
            const int r = tid >> 2, c = (tid & 3) * 8;
            const float* ap = A + (size_t)(m0 + r) * lda + (k0 + c);
            float4 f0 = *(const float4*)ap;
            float4 f1 = *(const float4*)(ap + 4);
            __bf16 t[8] = {(__bf16)f0.x, (__bf16)f0.y, (__bf16)f0.z, (__bf16)f0.w,
                           (__bf16)f1.x, (__bf16)f1.y, (__bf16)f1.z, (__bf16)f1.w};
            *(uint4*)&sA[r][c] = *(const uint4*)t;
        } else {
            for (int i = tid; i < BM * BK; i += 256) {
                int r = i >> 5, c = i & 31;
                int gr = m0 + r, gc = k0 + c;
                float v = (gr < M && gc < K) ? A[(size_t)gr * lda + gc] : 0.f;
                sA[r][c] = (__bf16)v;
            }
        }

        // ---- stage B tile (32x128), transposed into sBt[n][k] ----
        if (nFull && kFull) {
            // thread -> column (tid&127), 16 consecutive K rows; coalesced b32
            // loads, two b128 stores of the transposed strip
            const int c  = tid & 127;
            const int rb = (tid >> 7) * 16;
            const float* bp = B + (size_t)(k0 + rb) * ldb + (n0 + c);
            __bf16 t[16];
            #pragma unroll
            for (int r = 0; r < 16; ++r) {
                t[r] = (__bf16)(*bp);
                bp += ldb;
            }
            *(uint4*)&sBt[c][rb]     = *(const uint4*)&t[0];
            *(uint4*)&sBt[c][rb + 8] = *(const uint4*)&t[8];
        } else {
            for (int i = tid; i < BK * BN; i += 256) {
                int r = i >> 7, c = i & 127;
                int gr = k0 + r, gc = n0 + c;
                float v = (gr < K && gc < N) ? B[(size_t)gr * ldb + gc] : 0.f;
                sBt[c][r] = (__bf16)v;
            }
        }
        __syncthreads();

        // ---- fragments: two ds_load_b128 each ----
        union Frag { v16bf v; uint4 u[2]; };
        Frag a0, a1, b0, b1;
        const int ar = wm * 32 + (lane & 15);
        const int ak = (lane < 16) ? 0 : 8;
        a0.u[0] = *(const uint4*)&sA[ar][ak];
        a0.u[1] = *(const uint4*)&sA[ar][ak + 16];
        a1.u[0] = *(const uint4*)&sA[ar + 16][ak];
        a1.u[1] = *(const uint4*)&sA[ar + 16][ak + 16];
        const int bc = wn * 32 + (lane & 15);
        const int bk = (lane < 16) ? 0 : 8;
        b0.u[0] = *(const uint4*)&sBt[bc][bk];
        b0.u[1] = *(const uint4*)&sBt[bc][bk + 16];
        b1.u[0] = *(const uint4*)&sBt[bc + 16][bk];
        b1.u[1] = *(const uint4*)&sBt[bc + 16][bk + 16];

        acc00 = __builtin_amdgcn_wmma_f32_16x16x32_bf16(false, a0.v, false, b0.v,
                                                        (short)0, acc00, false, false);
        acc01 = __builtin_amdgcn_wmma_f32_16x16x32_bf16(false, a0.v, false, b1.v,
                                                        (short)0, acc01, false, false);
        acc10 = __builtin_amdgcn_wmma_f32_16x16x32_bf16(false, a1.v, false, b0.v,
                                                        (short)0, acc10, false, false);
        acc11 = __builtin_amdgcn_wmma_f32_16x16x32_bf16(false, a1.v, false, b1.v,
                                                        (short)0, acc11, false, false);
        __syncthreads();
    }

    // ---- epilogue ----
    const int crow = (lane < 16) ? 0 : 8;
    const int ccol = lane & 15;
    v8f* accs[4] = { &acc00, &acc01, &acc10, &acc11 };
    #pragma unroll
    for (int mt = 0; mt < 2; ++mt) {
        #pragma unroll
        for (int nt = 0; nt < 2; ++nt) {
            v8f& acc = *accs[mt * 2 + nt];
            const int gc = n0 + wn * 32 + nt * 16 + ccol;
            #pragma unroll
            for (int j = 0; j < 8; ++j) {
                const int gr = m0 + wm * 32 + mt * 16 + crow + j;
                if (gr < M && gc < N)
                    C[(size_t)gr * ldc + gc] = acc[j] + (bias ? bias[gc] : 0.f);
            }
        }
    }
}

// ---------------------------------------------------------------------------
// LayerNorm(row of 512) + ReLU, in place. One block per row.
// ---------------------------------------------------------------------------
__global__ __launch_bounds__(256) void ln_relu512(float* __restrict__ X,
                                                  const float* __restrict__ g,
                                                  const float* __restrict__ b) {
    int row = blockIdx.x, tid = threadIdx.x;
    float* xr = X + (size_t)row * 512;
    float x0 = xr[tid], x1 = xr[tid + 256];
    float mean = blockReduceSum256(x0 + x1) * (1.0f / 512.0f);
    float d0 = x0 - mean, d1 = x1 - mean;
    float var = blockReduceSum256(d0 * d0 + d1 * d1) * (1.0f / 512.0f);
    float inv = rsqrtf(var + 1e-5f);
    xr[tid]       = fmaxf(d0 * inv * g[tid] + b[tid], 0.f);
    xr[tid + 256] = fmaxf(d1 * inv * g[tid + 256] + b[tid + 256], 0.f);
}

// gate = sigmoid([r1|r2|attn] @ Wg + bg); out = gate*r1 + (1-gate)*r2
__global__ __launch_bounds__(256) void gate_combine(
    const float* __restrict__ r1, const float* __restrict__ r2,
    const float* __restrict__ attn, const float* __restrict__ Wg,
    const float* __restrict__ bg, float* __restrict__ out) {
    int row = blockIdx.x, tid = threadIdx.x;
    size_t o = (size_t)row * 512;
    float p = r1[o + tid] * Wg[tid] + r1[o + tid + 256] * Wg[tid + 256]
            + r2[o + tid] * Wg[512 + tid] + r2[o + tid + 256] * Wg[512 + tid + 256]
            + attn[o + tid] * Wg[1024 + tid] + attn[o + tid + 256] * Wg[1024 + tid + 256];
    float s = blockReduceSum256(p);
    float gate = 1.f / (1.f + expf(-(s + bg[0])));
    out[o + tid]       = gate * r1[o + tid]       + (1.f - gate) * r2[o + tid];
    out[o + tid + 256] = gate * r1[o + tid + 256] + (1.f - gate) * r2[o + tid + 256];
}

// ---------------------------------------------------------------------------
// Small utility kernels
// ---------------------------------------------------------------------------
__global__ void fill_f32(float* p, float v, int n) {
    int i = blockIdx.x * blockDim.x + threadIdx.x;
    if (i < n) p[i] = v;
}
__global__ void deg_count(float* deg, const int* __restrict__ dst, int E) {
    int e = blockIdx.x * blockDim.x + threadIdx.x;
    if (e < E) atomicAdd(&deg[dst[e]], 1.0f);
}

// out[i,c] = gcn_bias[c] + gat_bias[c] + xw[i,c] * (1/max(deg[i],1))  (self loop)
__global__ __launch_bounds__(256) void gcn_init(float* __restrict__ out,
    const float* __restrict__ xw, const float* __restrict__ deg,
    const float* __restrict__ b1, const float* __restrict__ b2) {
    int i = blockIdx.x, c = threadIdx.x;
    float dinv2 = 1.0f / fmaxf(deg[i], 1.0f);
    out[(size_t)i * OUTC + c] = b1[c] + b2[c] + xw[(size_t)i * OUTC + c] * dinv2;
}
__global__ __launch_bounds__(256) void gcn_scatter(float* __restrict__ out,
    const float* __restrict__ xw, const float* __restrict__ deg,
    const int* __restrict__ src, const int* __restrict__ dst) {
    int e = blockIdx.x, c = threadIdx.x;
    int s = src[e], d = dst[e];
    float norm = rsqrtf(fmaxf(deg[s], 1.f)) * rsqrtf(fmaxf(deg[d], 1.f));
    atomicAdd(&out[(size_t)d * OUTC + c], xw[(size_t)s * OUTC + c] * norm);
}

// ---------------------------------------------------------------------------
// GATv2 edge softmax
// ---------------------------------------------------------------------------
__global__ void gat_init(float* emax, float* eden, int n) {
    int i = blockIdx.x * blockDim.x + threadIdx.x;
    if (i < n) { emax[i] = -INFINITY; eden[i] = 0.f; }
}
__global__ __launch_bounds__(256) void gat_score(
    const float* __restrict__ xl, const float* __restrict__ xr,
    const float* __restrict__ att, const int* __restrict__ src,
    const int* __restrict__ dst, float* __restrict__ escore, float* emax) {
    int e = blockIdx.x, c = threadIdx.x;
    int s = src[e], d = dst[e];
    float v = xl[(size_t)s * OUTC + c] + xr[(size_t)d * OUTC + c];
    v = (v > 0.f) ? v : 0.2f * v;                 // leaky_relu(., 0.2)
    float sc = blockReduceSum256(v * att[c]);
    if (c == 0) { escore[e] = sc; atomicMaxF(&emax[d], sc); }
}
__global__ void gat_fixmax(float* emax, int n) {
    int i = blockIdx.x * blockDim.x + threadIdx.x;
    if (i < n && !__builtin_isfinite(emax[i])) emax[i] = 0.f;
}
__global__ void gat_exp(float* escore, const float* __restrict__ emax,
                        float* eden, const int* __restrict__ dst, int E) {
    int e = blockIdx.x * blockDim.x + threadIdx.x;
    if (e < E) {
        float ex = expf(escore[e] - emax[dst[e]]);
        escore[e] = ex;
        atomicAdd(&eden[dst[e]], ex);
    }
}
__global__ __launch_bounds__(256) void gat_scatter(float* __restrict__ out,
    const float* __restrict__ xl, const float* __restrict__ escore,
    const float* __restrict__ eden, const int* __restrict__ src,
    const int* __restrict__ dst) {
    int e = blockIdx.x, c = threadIdx.x;
    int s = src[e], d = dst[e];
    float alpha = escore[e] / fmaxf(eden[d], 1e-16f);
    atomicAdd(&out[(size_t)d * OUTC + c], xl[(size_t)s * OUTC + c] * alpha);
}

// ---------------------------------------------------------------------------
// BatchNorm (batch stats) + ReLU + SE
// ---------------------------------------------------------------------------
__global__ __launch_bounds__(256) void bn_stats(const float* __restrict__ x,
                                                float* sum, float* sumsq) {
    int i = blockIdx.x, c = threadIdx.x;
    float v = x[(size_t)i * OUTC + c];
    atomicAdd(&sum[c], v);
    atomicAdd(&sumsq[c], v * v);
}
__global__ __launch_bounds__(256) void bn_apply(const float* __restrict__ x,
    float* __restrict__ out, const float* __restrict__ sum,
    const float* __restrict__ sumsq, const float* __restrict__ g,
    const float* __restrict__ b, float* seSum, int n) {
    int i = blockIdx.x, c = threadIdx.x;
    float m   = sum[c] / (float)n;
    float var = sumsq[c] / (float)n - m * m;
    float y = fmaxf((x[(size_t)i * OUTC + c] - m) * rsqrtf(var + 1e-5f) * g[c] + b[c], 0.f);
    out[(size_t)i * OUTC + c] = y;
    atomicAdd(&seSum[c], y);
}
__global__ __launch_bounds__(256) void se_gate(const float* __restrict__ seSum,
    const float* __restrict__ W1, const float* __restrict__ W2,
    float* __restrict__ y, int n) {
    __shared__ float mean[256];
    __shared__ float t[16];
    int c = threadIdx.x;
    mean[c] = seSum[c] / (float)n;
    __syncthreads();
    if (c < 16) {
        float a = 0.f;
        for (int k = 0; k < 256; ++k) a += mean[k] * W1[k * 16 + c];
        t[c] = fmaxf(a, 0.f);
    }
    __syncthreads();
    float a = 0.f;
    #pragma unroll
    for (int j = 0; j < 16; ++j) a += t[j] * W2[j * 256 + c];
    y[c] = 1.f / (1.f + expf(-a));
}
__global__ __launch_bounds__(256) void se_apply(float* __restrict__ x,
                                                const float* __restrict__ y) {
    int i = blockIdx.x, c = threadIdx.x;
    x[(size_t)i * OUTC + c] *= y[c];
}

__global__ __launch_bounds__(256) void resid_out(const float* __restrict__ res,
    const float* __restrict__ x, float* __restrict__ out) {
    int i = blockIdx.x, c = threadIdx.x;
    out[(size_t)i * OUTC + c] = res[(size_t)i * OUTC + c] + x[(size_t)i * OUTC + c];
}

// conbs[i] = [xd_final[dis] | xg_final[gene]]  (pos first, then neg)
__global__ __launch_bounds__(256) void build_pairs(const float* __restrict__ xd,
    const float* __restrict__ xg, const int* __restrict__ pos,
    const int* __restrict__ neg, float* __restrict__ conbs) {
    int i = blockIdx.x, c = threadIdx.x;
    const int* e = (i < 8192) ? (pos + 2 * i) : (neg + 2 * (i - 8192));
    int dis = e[0], gen = e[1];
    conbs[(size_t)i * 512 + c]       = xd[(size_t)dis * OUTC + c];
    conbs[(size_t)i * 512 + 256 + c] = xg[(size_t)gen * OUTC + c];
}

// p[i] = sigmoid(relu(h[i]) . W2 + b2); terms[i] = BCE(p[i], target_i)
__global__ __launch_bounds__(256) void head(const float* __restrict__ h,
    const float* __restrict__ W2, const float* __restrict__ b2,
    float* __restrict__ pout, float* __restrict__ terms) {
    int i = blockIdx.x, c = threadIdx.x;
    float v = fmaxf(h[(size_t)i * OUTC + c], 0.f) * W2[c];
    float s = blockReduceSum256(v);
    if (c == 0) {
        float p = 1.f / (1.f + expf(-(s + b2[0])));
        pout[i] = p;
        float pc = fminf(fmaxf(p, 1e-7f), 1.f - 1e-7f);
        float t = (i < 8192) ? 1.f : 0.f;
        terms[i] = -(t * logf(pc) + (1.f - t) * logf(1.f - pc));
    }
}
__global__ __launch_bounds__(256) void loss_reduce(const float* __restrict__ terms,
                                                   float* __restrict__ out, int n) {
    __shared__ float red[256];
    int tid = threadIdx.x;
    float s = 0.f;
    for (int i = tid; i < n; i += 256) s += terms[i];
    red[tid] = s;
    __syncthreads();
    for (int k = 128; k > 0; k >>= 1) {
        if (tid < k) red[tid] += red[tid + k];
        __syncthreads();
    }
    if (tid == 0) out[0] = red[0] / (float)n;
}

// ---------------------------------------------------------------------------
// Host side
// ---------------------------------------------------------------------------
static void gemmF(hipStream_t st, const float* A, int lda, const float* B, int ldb,
                  const float* bias, float* C, int ldc, int M, int N, int K) {
    dim3 g((N + BN - 1) / BN, (M + BM - 1) / BM), b(256);
    gemm_bf16_wmma<<<g, b, 0, st>>>(A, lda, B, ldb, bias, C, ldc, M, N, K);
}

struct Gate { const float *W1,*W2,*Wg,*Wo,*Wv,*b1,*b2,*bg,*bo,*bv,*ln1b,*ln1g,*ln2b,*ln2g; };
struct Layer { const float *bnb,*bng,*dd_W,*dd_b,
               *dgWl,*dgWr,*dgatt,*dgbias,*dgbl,*dgbr,
               *gdWl,*gdWr,*gdatt,*gdbias,*gdbl,*gdbr,
               *gg_W,*gg_b,*seW1,*seW2; };

extern "C" void kernel_launch(void* const* d_in, const int* in_sizes, int n_in,
                              void* d_out, int out_size, void* d_ws, size_t ws_size,
                              hipStream_t stream) {
    (void)in_sizes; (void)n_in; (void)out_size; (void)ws_size;

    // ---- input layout: top-level insertion order; nested params dict flattened
    //      jax-style (sorted keys at each level) ----
    const float* x_gene = (const float*)d_in[0];   // [12000, 6512]
    const float* x_dis  = (const float*)d_in[1];   // [6000, 12512]

    auto F = [&](int i) { return (const float*)d_in[i]; };
    auto loadGate = [&](int b) {
        Gate g;
        g.W1 = F(b+0);  g.W2 = F(b+1);  g.Wg = F(b+2);  g.Wo = F(b+3);  g.Wv = F(b+4);
        g.b1 = F(b+5);  g.b2 = F(b+6);  g.bg = F(b+7);  g.bo = F(b+8);  g.bv = F(b+9);
        g.ln1b = F(b+10); g.ln1g = F(b+11); g.ln2b = F(b+12); g.ln2g = F(b+13);
        return g;
    };
    Gate dG = loadGate(2);                 // d_gate
    const float* d_lin_W = F(16);
    const float* d_lin_b = F(17);
    Gate gG = loadGate(18);                // g_gate
    const float* g_lin_W = F(32);
    const float* g_lin_b = F(33);
    auto loadLayer = [&](int b) {
        Layer L;
        L.bnb = F(b+0);   L.bng = F(b+1);   L.dd_W = F(b+2);  L.dd_b = F(b+3);
        L.dgWl = F(b+4);  L.dgWr = F(b+5);  L.dgatt = F(b+6); L.dgbias = F(b+7);
        L.dgbl = F(b+8);  L.dgbr = F(b+9);
        L.gdWl = F(b+10); L.gdWr = F(b+11); L.gdatt = F(b+12); L.gdbias = F(b+13);
        L.gdbl = F(b+14); L.gdbr = F(b+15);
        L.gg_W = F(b+16); L.gg_b = F(b+17); L.seW1 = F(b+18); L.seW2 = F(b+19);
        return L;
    };
    Layer L0 = loadLayer(34), L1 = loadLayer(54);
    const float* mlp_W1 = F(74);           // [512,256]
    const float* mlp_W2 = F(75);           // [256,1]
    const float* mlp_b1 = F(76);
    const float* mlp_b2 = F(77);
    const int* ei_gg = (const int*)d_in[78];  // [2, 384000]
    const int* ei_dd = (const int*)d_in[79];  // [2, 192000]
    const int* ei_dg = (const int*)d_in[80];  // [2, 300000] src=disease dst=gene
    const int* ei_gd = (const int*)d_in[81];  // [2, 300000] src=gene dst=disease
    const int* pose  = (const int*)d_in[82];  // [8192, 2] (disease, gene)
    const int* nege  = (const int*)d_in[83];

    // ---- workspace layout ----
    float* W = (float*)d_ws;
    size_t o = 0;
    auto take = [&](size_t n) { float* p = W + o; o += n; return p; };
    float* buf0 = take((size_t)G_N * HID);   // also: conbs[16384x512] spans buf0+buf1
    float* buf1 = take((size_t)G_N * HID);
    float* buf2 = take((size_t)G_N * HID);   // also: h[16384x256]
    float* buf3 = take((size_t)G_N * HID);
    float* resg = take((size_t)G_N * OUTC);
    float* resd = take((size_t)D_N * OUTC);
    float* xgw  = take((size_t)G_N * OUTC);
    float* xdw  = take((size_t)D_N * OUTC);
    float* gnew = take((size_t)G_N * OUTC);
    float* dnew = take((size_t)D_N * OUTC);
    float* degg = take(G_N);
    float* degd = take(D_N);
    float* emax = take(G_N);
    float* eden = take(G_N);
    float* escore = take(E_DG);              // also: BCE terms (16384)
    float* statSum = take(256);
    float* statSq  = take(256);
    float* seSum   = take(256);
    float* ygate   = take(256);
    float* conbs = buf0;
    float* hbuf  = buf2;

    // ---- output layout: [loss(1) | p(16384) | xg(12000x256) | xd(6000x256)] ----
    float* out    = (float*)d_out;
    float* outP   = out + 1;
    float* outXg  = out + 1 + NPAIR;
    float* outXd  = outXg + (size_t)G_N * OUTC;

    // ======================= gating (gene then disease) =======================
    auto run_gating = [&](const float* X, int ldx, int relW, const Gate& gp, int n,
                          const float* linW, const float* linb, float* resOut) {
        gemmF(stream, X, ldx, gp.W1, HID, gp.b1, buf0, HID, n, HID, 512);        // r1
        ln_relu512<<<n, 256, 0, stream>>>(buf0, gp.ln1g, gp.ln1b);
        gemmF(stream, X + 512, ldx, gp.W2, HID, gp.b2, buf1, HID, n, HID, relW); // r2
        ln_relu512<<<n, 256, 0, stream>>>(buf1, gp.ln2g, gp.ln2b);
        gemmF(stream, buf1, HID, gp.Wv, HID, gp.bv, buf2, HID, n, HID, HID);     // v
        gemmF(stream, buf2, HID, gp.Wo, HID, gp.bo, buf3, HID, n, HID, HID);     // attn
        gate_combine<<<n, 256, 0, stream>>>(buf0, buf1, buf3, gp.Wg, gp.bg, buf2);
        gemmF(stream, buf2, HID, linW, OUTC, linb, resOut, OUTC, n, OUTC, HID);
    };
    run_gating(x_gene, 512 + D_N, D_N, gG, G_N, g_lin_W, g_lin_b, resg);
    run_gating(x_dis,  512 + G_N, G_N, dG, D_N, d_lin_W, d_lin_b, resd);

    hipMemcpyAsync(xgw, resg, (size_t)G_N * OUTC * sizeof(float),
                   hipMemcpyDeviceToDevice, stream);
    hipMemcpyAsync(xdw, resd, (size_t)D_N * OUTC * sizeof(float),
                   hipMemcpyDeviceToDevice, stream);

    // ======================= 2 hetero conv layers =======================
    const Layer* Ls[2] = { &L0, &L1 };
    for (int l = 0; l < 2; ++l) {
        const Layer& L = *Ls[l];

        // ---- gene side: gnew = GCN(gg) + GATv2(dg: disease->gene) ----
        gemmF(stream, xgw, OUTC, L.gg_W, OUTC, nullptr, buf0, OUTC, G_N, OUTC, OUTC);
        fill_f32<<<(G_N + 255) / 256, 256, 0, stream>>>(degg, 1.0f, G_N);
        deg_count<<<(E_GG + 255) / 256, 256, 0, stream>>>(degg, ei_gg + E_GG, E_GG);
        gcn_init<<<G_N, 256, 0, stream>>>(gnew, buf0, degg, L.gg_b, L.dgbias);
        gcn_scatter<<<E_GG, 256, 0, stream>>>(gnew, buf0, degg, ei_gg, ei_gg + E_GG);

        gemmF(stream, xdw, OUTC, L.dgWl, OUTC, L.dgbl, buf1, OUTC, D_N, OUTC, OUTC);
        gemmF(stream, xgw, OUTC, L.dgWr, OUTC, L.dgbr, buf2, OUTC, G_N, OUTC, OUTC);
        gat_init<<<(G_N + 255) / 256, 256, 0, stream>>>(emax, eden, G_N);
        gat_score<<<E_DG, 256, 0, stream>>>(buf1, buf2, L.dgatt, ei_dg, ei_dg + E_DG,
                                            escore, emax);
        gat_fixmax<<<(G_N + 255) / 256, 256, 0, stream>>>(emax, G_N);
        gat_exp<<<(E_DG + 255) / 256, 256, 0, stream>>>(escore, emax, eden,
                                                        ei_dg + E_DG, E_DG);
        gat_scatter<<<E_DG, 256, 0, stream>>>(gnew, buf1, escore, eden,
                                              ei_dg, ei_dg + E_DG);

        // ---- disease side: dnew = GCN(dd) + GATv2(gd: gene->disease) ----
        gemmF(stream, xdw, OUTC, L.dd_W, OUTC, nullptr, buf0, OUTC, D_N, OUTC, OUTC);
        fill_f32<<<(D_N + 255) / 256, 256, 0, stream>>>(degd, 1.0f, D_N);
        deg_count<<<(E_DD + 255) / 256, 256, 0, stream>>>(degd, ei_dd + E_DD, E_DD);
        gcn_init<<<D_N, 256, 0, stream>>>(dnew, buf0, degd, L.dd_b, L.gdbias);
        gcn_scatter<<<E_DD, 256, 0, stream>>>(dnew, buf0, degd, ei_dd, ei_dd + E_DD);

        gemmF(stream, xgw, OUTC, L.gdWl, OUTC, L.gdbl, buf1, OUTC, G_N, OUTC, OUTC);
        gemmF(stream, xdw, OUTC, L.gdWr, OUTC, L.gdbr, buf2, OUTC, D_N, OUTC, OUTC);
        gat_init<<<(D_N + 255) / 256, 256, 0, stream>>>(emax, eden, D_N);
        gat_score<<<E_GD, 256, 0, stream>>>(buf1, buf2, L.gdatt, ei_gd, ei_gd + E_GD,
                                            escore, emax);
        gat_fixmax<<<(D_N + 255) / 256, 256, 0, stream>>>(emax, D_N);
        gat_exp<<<(E_GD + 255) / 256, 256, 0, stream>>>(escore, emax, eden,
                                                        ei_gd + E_GD, E_GD);
        gat_scatter<<<E_GD, 256, 0, stream>>>(dnew, buf1, escore, eden,
                                              ei_gd, ei_gd + E_GD);

        // ---- BN + ReLU + SE, updating xgw / xdw ----
        fill_f32<<<3, 256, 0, stream>>>(statSum, 0.f, 768);
        bn_stats<<<G_N, 256, 0, stream>>>(gnew, statSum, statSq);
        bn_apply<<<G_N, 256, 0, stream>>>(gnew, xgw, statSum, statSq,
                                          L.bng, L.bnb, seSum, G_N);
        se_gate<<<1, 256, 0, stream>>>(seSum, L.seW1, L.seW2, ygate, G_N);
        se_apply<<<G_N, 256, 0, stream>>>(xgw, ygate);

        fill_f32<<<3, 256, 0, stream>>>(statSum, 0.f, 768);
        bn_stats<<<D_N, 256, 0, stream>>>(dnew, statSum, statSq);
        bn_apply<<<D_N, 256, 0, stream>>>(dnew, xdw, statSum, statSq,
                                          L.bng, L.bnb, seSum, D_N);
        se_gate<<<1, 256, 0, stream>>>(seSum, L.seW1, L.seW2, ygate, D_N);
        se_apply<<<D_N, 256, 0, stream>>>(xdw, ygate);
    }

    // ======================= residual + link predictor =======================
    resid_out<<<G_N, 256, 0, stream>>>(resg, xgw, outXg);
    resid_out<<<D_N, 256, 0, stream>>>(resd, xdw, outXd);

    build_pairs<<<NPAIR, 256, 0, stream>>>(outXd, outXg, pose, nege, conbs);
    gemmF(stream, conbs, 512, mlp_W1, OUTC, mlp_b1, hbuf, OUTC, NPAIR, OUTC, 512);
    head<<<NPAIR, 256, 0, stream>>>(hbuf, mlp_W2, mlp_b2, outP, escore);
    loss_reduce<<<1, 256, 0, stream>>>(escore, out, NPAIR);
}
